// GRUCell_21878563406418
// MI455X (gfx1250) — compile-verified, matrix operand order checked by reading, not run on hardware
//
#include <hip/hip_runtime.h>

// Problem constants
#define IN_ 512
#define H_  1024
#define B_  64
#define S_  1024

typedef __attribute__((ext_vector_type(16))) __bf16 v16bf;
typedef __attribute__((ext_vector_type(8)))  float  v8f;

__device__ __forceinline__ int frag_k(int lane, int j) {
    // 16-bit A/B fragment k-mapping (CDNA5 ISA 7.12.2):
    // element j: k = (j>>3)*16 + (lane>>4)*8 + (j&7)
    return ((j >> 3) << 4) + ((lane >> 4) << 3) + (j & 7);
}

__device__ __forceinline__ float sigm(float x) { return 1.0f / (1.0f + __expf(-x)); }

#define WMMA_BF16(A, Bv, C) \
    __builtin_amdgcn_wmma_f32_16x16x32_bf16(false, (A), false, (Bv), (short)0, (C), false, false)

// ---------------- weight pre-swizzle into B-fragment blob order ----------------
// Blob = (ktile*NTILES + ntile): 32 lanes x 16 bf16, lane-contiguous (32B/lane).

__global__ void pack_gates_w(const float* __restrict__ Wr, const float* __restrict__ Wz,
                             const float* __restrict__ Ur, const float* __restrict__ Uz,
                             __bf16* __restrict__ Wg) {
    int idx = blockIdx.x * 256 + threadIdx.x;           // < 1536*2048
    int j = idx & 15, lane = (idx >> 4) & 31, blob = idx >> 9;
    int ktile = blob >> 7, ntile = blob & 127;          // 48 x 128 blobs
    int k = (ktile << 5) + frag_k(lane, j);
    int n = (ntile << 4) + (lane & 15);
    float v;
    if (k < IN_) v = (n < H_) ? Wr[k * H_ + n] : Wz[k * H_ + (n - H_)];
    else { int kk = k - IN_; v = (n < H_) ? Ur[kk * H_ + n] : Uz[kk * H_ + (n - H_)]; }
    Wg[idx] = (__bf16)v;
}

__global__ void pack_s_w(const float* __restrict__ Ws, const float* __restrict__ Us,
                         __bf16* __restrict__ Wsf) {
    int idx = blockIdx.x * 256 + threadIdx.x;           // < 1536*1024
    int j = idx & 15, lane = (idx >> 4) & 31, blob = idx >> 9;
    int ktile = blob >> 6, ntile = blob & 63;           // 48 x 64 blobs
    int k = (ktile << 5) + frag_k(lane, j);
    int n = (ntile << 4) + (lane & 15);
    float v = (k < IN_) ? Ws[k * H_ + n] : Us[(k - IN_) * H_ + n];
    Wsf[idx] = (__bf16)v;
}

__global__ void pack_o_w(const float* __restrict__ Wo, __bf16* __restrict__ Wof) {
    int idx = blockIdx.x * 256 + threadIdx.x;           // < 1024*512
    int j = idx & 15, lane = (idx >> 4) & 31, blob = idx >> 9;
    int ktile = blob >> 5, ntile = blob & 31;           // 32 x 32 blobs
    int k = (ktile << 5) + frag_k(lane, j);
    int n = (ntile << 4) + (lane & 15);
    Wof[idx] = (__bf16)Wo[k * IN_ + n];
}

__global__ void init_h(const float* __restrict__ h0, float* __restrict__ h) {
    int i = blockIdx.x * 256 + threadIdx.x;             // < 64*1024
    h[i] = h0[i & (H_ - 1)];
}

// ---------------- per-step kernel 1: r,z gates ----------------
// D[64,2048] = [x_t | h][64,1536] @ Wg[1536,2048]; r = sigmoid(D[:, :1024]+br),
// z = sigmoid(D[:, 1024:]+bz); emit rh = r*h (bf16) and z (f32).
// Grid: 8 WGs x 256 thr. WG tile 64 x 256. Wave tile: 2 mtiles x 4 ntiles.
__global__ void __launch_bounds__(256) gru_gates(
    const float* __restrict__ x, const __bf16* __restrict__ Wg,
    const float* __restrict__ br, const float* __restrict__ bz,
    const float* __restrict__ h, __bf16* __restrict__ rh,
    float* __restrict__ z, int t)
{
    __shared__ __align__(32) __bf16 ldsA[16384];        // 64 rows x 256 k (32 KB)
    const int tid = threadIdx.x;
    const int L = tid & 31, w = tid >> 5;
    const int rw = w & 1, cw = w >> 1;                  // 2 row-groups x 4 col-groups

    // per-thread staging constants: j = tid&15, lanes tid>>4 and tid>>4 + 16
    const int tj = tid & 15, tl = tid >> 4;
    const int kj = ((tj >> 3) << 4) + (tj & 7);

    // wave-invariant base pointers
    const __bf16* bp = Wg + (((size_t)(blockIdx.x * 16 + cw * 4)) << 9) + (L << 4);
    const __bf16* ap = &ldsA[(rw << 13) + (L << 4)];    // rw*16 blobs * 512

    v8f acc[2][4] = {};
    for (int kc = 0; kc < 6; ++kc) {                    // 6 chunks of 256 k
        if (kc) __syncthreads();
        // ---- stage A chunk (64 x 256) in fragment order; x for kc<2, h else ----
#pragma unroll
        for (int mt = 0; mt < 4; ++mt) {
            const int m = (mt << 4) + tl;
            __bf16* dst = &ldsA[(mt << 12) + (tl << 4) + tj];   // blob = mt*8 + kb
            if (kc < 2) {
                const float* xr = x + (size_t)m * (S_ * IN_) + (size_t)t * IN_ + kc * 256 + kj;
#pragma unroll
                for (int kb = 0; kb < 8; ++kb) {
                    dst[kb * 512]       = (__bf16)xr[kb * 32];
                    dst[kb * 512 + 256] = (__bf16)xr[kb * 32 + 8];
                }
            } else {
                const float* hr = h + m * H_ + (kc - 2) * 256 + kj;
#pragma unroll
                for (int kb = 0; kb < 8; ++kb) {
                    dst[kb * 512]       = (__bf16)hr[kb * 32];
                    dst[kb * 512 + 256] = (__bf16)hr[kb * 32 + 8];
                }
            }
        }
        __syncthreads();
        // ---- 8 ktiles: 2 A frags + 4 B frags -> 8 WMMAs, all immediate offsets ----
#pragma unroll
        for (int kt = 0; kt < 8; ++kt) {
            v16bf a0 = *(const v16bf*)(ap + kt * 512);
            v16bf a1 = *(const v16bf*)(ap + 4096 + kt * 512);
#pragma unroll
            for (int i = 0; i < 4; ++i) {
                v16bf b = *(const v16bf*)(bp + (kt * 128 + i) * 512);
                acc[0][i] = WMMA_BF16(a0, b, acc[0][i]);
                acc[1][i] = WMMA_BF16(a1, b, acc[1][i]);
            }
        }
        bp += 8 * 128 * 512;                            // next K chunk of blobs
    }
    // ---- epilogue: D layout m = v + 8*(L>>4), n = L&15 ----
#pragma unroll
    for (int mi = 0; mi < 2; ++mi) {
#pragma unroll
        for (int i = 0; i < 4; ++i) {
            int n = (blockIdx.x << 8) + ((cw * 4 + i) << 4) + (L & 15);
#pragma unroll
            for (int v = 0; v < 8; ++v) {
                int m = ((2 * rw + mi) << 4) + ((L >> 4) << 3) + v;
                float pre = acc[mi][i][v];
                if (n < H_) {
                    float r = sigm(pre + br[n]);
                    rh[m * H_ + n] = (__bf16)(r * h[m * H_ + n]);
                } else {
                    int nz = n - H_;
                    z[m * H_ + nz] = sigm(pre + bz[nz]);
                }
            }
        }
    }
}

// ---------------- per-step kernel 2: candidate + state update ----------------
// s = tanh([x_t | rh] @ Wsf + bs); h = (1-z)h + z*s; log h into h_all (bf16).
// Grid: 4 WGs x 256 thr. WG tile 64 x 256. Wave tile 2m x 4n.
__global__ void __launch_bounds__(256) gru_hnew(
    const float* __restrict__ x, const __bf16* __restrict__ Wsf,
    const float* __restrict__ bs, const __bf16* __restrict__ rh,
    const float* __restrict__ zbuf, float* __restrict__ h,
    __bf16* __restrict__ h_all, int t)
{
    __shared__ __align__(32) __bf16 ldsA[16384];
    const int tid = threadIdx.x;
    const int L = tid & 31, w = tid >> 5;
    const int rw = w & 1, cw = w >> 1;
    const int tj = tid & 15, tl = tid >> 4;
    const int kj = ((tj >> 3) << 4) + (tj & 7);

    const __bf16* bp = Wsf + (((size_t)(blockIdx.x * 16 + cw * 4)) << 9) + (L << 4);
    const __bf16* ap = &ldsA[(rw << 13) + (L << 4)];

    v8f acc[2][4] = {};
    for (int kc = 0; kc < 6; ++kc) {
        if (kc) __syncthreads();
#pragma unroll
        for (int mt = 0; mt < 4; ++mt) {
            const int m = (mt << 4) + tl;
            __bf16* dst = &ldsA[(mt << 12) + (tl << 4) + tj];
            if (kc < 2) {
                const float* xr = x + (size_t)m * (S_ * IN_) + (size_t)t * IN_ + kc * 256 + kj;
#pragma unroll
                for (int kb = 0; kb < 8; ++kb) {
                    dst[kb * 512]       = (__bf16)xr[kb * 32];
                    dst[kb * 512 + 256] = (__bf16)xr[kb * 32 + 8];
                }
            } else {
                const __bf16* rr = rh + m * H_ + (kc - 2) * 256 + kj;
#pragma unroll
                for (int kb = 0; kb < 8; ++kb) {
                    dst[kb * 512]       = rr[kb * 32];
                    dst[kb * 512 + 256] = rr[kb * 32 + 8];
                }
            }
        }
        __syncthreads();
#pragma unroll
        for (int kt = 0; kt < 8; ++kt) {
            v16bf a0 = *(const v16bf*)(ap + kt * 512);
            v16bf a1 = *(const v16bf*)(ap + 4096 + kt * 512);
#pragma unroll
            for (int i = 0; i < 4; ++i) {
                v16bf b = *(const v16bf*)(bp + (kt * 64 + i) * 512);
                acc[0][i] = WMMA_BF16(a0, b, acc[0][i]);
                acc[1][i] = WMMA_BF16(a1, b, acc[1][i]);
            }
        }
        bp += 8 * 64 * 512;
    }
#pragma unroll
    for (int mi = 0; mi < 2; ++mi) {
#pragma unroll
        for (int i = 0; i < 4; ++i) {
            int n = (blockIdx.x << 8) + ((cw * 4 + i) << 4) + (L & 15);
#pragma unroll
            for (int v = 0; v < 8; ++v) {
                int m = ((2 * rw + mi) << 4) + ((L >> 4) << 3) + v;
                int idx = m * H_ + n;
                float s  = tanhf(acc[mi][i][v] + bs[n]);
                float zv = zbuf[idx];
                float hn = h[idx] + zv * (s - h[idx]);  // (1-z)h + z*s
                h[idx] = hn;
                h_all[((size_t)m * S_ + t) * H_ + n] = (__bf16)hn;
            }
        }
    }
}

// ---------------- deferred output: softmax(h_all @ Wo + bo) ----------------
// One WG per 32 GEMM rows (row = b*S + t). Wave tile 2m x 4n; logits in LDS
// (overlaid on the A stage buffer). Grid: 2048 WGs.
__global__ void __launch_bounds__(256) gru_out(
    const __bf16* __restrict__ h_all, const __bf16* __restrict__ Wof,
    const float* __restrict__ bo, float* __restrict__ y)
{
    __shared__ __align__(32) union { __bf16 a[32768]; float lg[16384]; } smem; // 64 KB
    __shared__ float red[32][9];
    const int tid = threadIdx.x, L = tid & 31, w = tid >> 5;
    const int tj = tid & 15, tl = tid >> 4;
    const int kj = ((tj >> 3) << 4) + (tj & 7);
    const size_t rbase = (size_t)blockIdx.x * 32;

    // ---- stage 32 x 1024 A panel in fragment order; blob = mt*32 + kt ----
#pragma unroll
    for (int mt = 0; mt < 2; ++mt) {
        const __bf16* rp = h_all + (rbase + (mt << 4) + tl) * H_ + kj;
        __bf16* dst = &smem.a[(mt << 14) + (tl << 4) + tj];
#pragma unroll 8
        for (int kt = 0; kt < 32; ++kt) {
            dst[kt * 512]       = rp[kt * 32];
            dst[kt * 512 + 256] = rp[kt * 32 + 8];
        }
    }
    __syncthreads();

    const __bf16* bp = Wof + (((size_t)(w * 4)) << 9) + (L << 4);
    v8f acc[2][4] = {};
    for (int kc = 0; kc < 4; ++kc) {                    // 4 chunks of 8 ktiles
        const __bf16* ap = &smem.a[(kc << 12) + (L << 4)];
#pragma unroll
        for (int kt = 0; kt < 8; ++kt) {
            v16bf a0 = *(const v16bf*)(ap + kt * 512);
            v16bf a1 = *(const v16bf*)(ap + 16384 + kt * 512);
#pragma unroll
            for (int i = 0; i < 4; ++i) {
                v16bf b = *(const v16bf*)(bp + (kt * 32 + i) * 512);
                acc[0][i] = WMMA_BF16(a0, b, acc[0][i]);
                acc[1][i] = WMMA_BF16(a1, b, acc[1][i]);
            }
        }
        bp += 8 * 32 * 512;
    }
    __syncthreads();                                    // done reading smem.a
#pragma unroll
    for (int mi = 0; mi < 2; ++mi) {
#pragma unroll
        for (int i = 0; i < 4; ++i) {
            int n = ((w * 4 + i) << 4) + (L & 15);
#pragma unroll
            for (int v = 0; v < 8; ++v) {
                int m = (mi << 4) + ((L >> 4) << 3) + v;
                smem.lg[m * IN_ + n] = acc[mi][i][v] + bo[n];
            }
        }
    }
    __syncthreads();
    // ---- softmax: 8 threads per row, 64 cols each ----
    int row = tid >> 3, seg = tid & 7;
    float mx = -3.4e38f;
    for (int c = seg * 64; c < seg * 64 + 64; ++c) mx = fmaxf(mx, smem.lg[row * IN_ + c]);
    red[row][seg] = mx;
    __syncthreads();
    mx = -3.4e38f;
    for (int s2 = 0; s2 < 8; ++s2) mx = fmaxf(mx, red[row][s2]);
    float sm = 0.0f;
    for (int c = seg * 64; c < seg * 64 + 64; ++c) sm += __expf(smem.lg[row * IN_ + c] - mx);
    __syncthreads();
    red[row][seg] = sm;
    __syncthreads();
    sm = 0.0f;
    for (int s2 = 0; s2 < 8; ++s2) sm += red[row][s2];
    float inv = 1.0f / sm;
    for (int c = seg * 64; c < seg * 64 + 64; ++c)
        y[(rbase + row) * IN_ + c] = __expf(smem.lg[row * IN_ + c] - mx) * inv;
}

__global__ void copy_h(const float* __restrict__ h, float* __restrict__ out) {
    int i = blockIdx.x * 256 + threadIdx.x;             // < 64*1024
    out[i] = h[i];
}

// ---------------- host-side orchestration ----------------
extern "C" void kernel_launch(void* const* d_in, const int* in_sizes, int n_in,
                              void* d_out, int out_size, void* d_ws, size_t ws_size,
                              hipStream_t stream) {
    const float* x  = (const float*)d_in[0];
    const float* Wr = (const float*)d_in[1];  const float* br = (const float*)d_in[2];
    const float* Ur = (const float*)d_in[3];
    const float* Wz = (const float*)d_in[4];  const float* bz = (const float*)d_in[5];
    const float* Uz = (const float*)d_in[6];
    const float* Ws = (const float*)d_in[7];  const float* bs = (const float*)d_in[8];
    const float* Us = (const float*)d_in[9];
    const float* Wo = (const float*)d_in[10]; const float* bo = (const float*)d_in[11];
    const float* h0 = (const float*)d_in[12];

    float* y     = (float*)d_out;                       // [B,S,IN] row = b*S+t
    float* h_out = y + (size_t)B_ * S_ * IN_;           // [B,H]

    // workspace layout (bytes): total ~138.6 MB
    char* ws = (char*)d_ws;
    __bf16* Wg    = (__bf16*)(ws);                              // 1536x2048 bf16 = 6 MB
    __bf16* Wsf   = (__bf16*)(ws + 6291456);                    // 1536x1024 bf16 = 3 MB
    __bf16* Wof   = (__bf16*)(ws + 9437184);                    // 1024x512  bf16 = 1 MB
    float*  hbuf  = (float*) (ws + 10485760);                   // 64x1024 f32
    __bf16* rhbuf = (__bf16*)(ws + 10747904);                   // 64x1024 bf16
    float*  zbuf  = (float*) (ws + 10878976);                   // 64x1024 f32
    __bf16* h_all = (__bf16*)(ws + 11141120);                   // 65536x1024 bf16 = 128 MB

    pack_gates_w<<<12288, 256, 0, stream>>>(Wr, Wz, Ur, Uz, Wg);
    pack_s_w    <<<6144,  256, 0, stream>>>(Ws, Us, Wsf);
    pack_o_w    <<<2048,  256, 0, stream>>>(Wo, Wof);
    init_h      <<<256,   256, 0, stream>>>(h0, hbuf);

    for (int t = 0; t < S_; ++t) {
        gru_gates<<<8, 256, 0, stream>>>(x, Wg, br, bz, hbuf, rhbuf, zbuf, t);
        gru_hnew <<<4, 256, 0, stream>>>(x, Wsf, bs, rhbuf, zbuf, hbuf, h_all, t);
    }

    gru_out<<<(B_ * S_) / 32, 256, 0, stream>>>(h_all, Wof, bo, y);
    copy_h <<<256, 256, 0, stream>>>(hbuf, h_out);
}
